// UNetAttention_89429809038118
// MI455X (gfx1250) — matrix operand order repaired and checked by reading, not compiled, not run on hardware
//
#include <hip/hip_runtime.h>
#include <hip/hip_bf16.h>
#include <stdint.h>

// ---------------------------------------------------------------------------
// UNet self-attention for MI455X (gfx1250, wave32, WMMA, double-buffered
// async-to-LDS staging).
// ---------------------------------------------------------------------------

typedef __bf16 bf16;
typedef __attribute__((ext_vector_type(8)))  __bf16 bf16x8;
typedef __attribute__((ext_vector_type(16))) __bf16 bf16x16;
typedef __attribute__((ext_vector_type(8)))  float  v8f;

#define CH    1280
#define SEQ   4096
#define BATCH 2
#define HEADS 8
#define HDIM  160

static __device__ inline v8f v8f_zero() {
  v8f z;
#pragma unroll
  for (int i = 0; i < 8; ++i) z[i] = 0.0f;
  return z;
}

static __device__ inline v8f wmma_bf16(bf16x16 a, bf16x16 b, v8f c) {
  return __builtin_amdgcn_wmma_f32_16x16x32_bf16(false, a, false, b, (short)0, c,
                                                 false, false);
}

// A-fragment (16x32 bf16): lane = row (lane&15); elems 0..7 = K (lane>>4)*8..+7,
// elems 8..15 = +16..+23 (ISA 7.12.2).
static __device__ inline bf16x16 load_afrag(const bf16* tile, int ld) {
  const int lane = threadIdx.x & 31;
  const bf16* p = tile + (size_t)(lane & 15) * ld + ((lane >> 4) << 3);
  bf16x8 lo = *(const bf16x8*)p;
  bf16x8 hi = *(const bf16x8*)(p + 16);
  bf16x16 f;
#pragma unroll
  for (int i = 0; i < 8; ++i) { f[i] = lo[i]; f[i + 8] = hi[i]; }
  return f;
}

// B-fragment (32x16 bf16) from [n][k] row-major source: lane = column n
// (lane&15); elems 0..7 = K (lane>>4)*8..+7, elems 8..15 = +16..+23.
static __device__ inline bf16x16 load_bfrag(const bf16* tile, int ld) {
  const int lane = threadIdx.x & 31;
  const bf16* p = tile + (size_t)(lane & 15) * ld + ((lane >> 4) << 3);
  bf16x8 lo = *(const bf16x8*)p;
  bf16x8 hi = *(const bf16x8*)(p + 16);
  bf16x16 f;
#pragma unroll
  for (int i = 0; i < 8; ++i) { f[i] = lo[i]; f[i + 8] = hi[i]; }
  return f;
}

// CDNA5 async copy: 16B global -> LDS, tracked by ASYNCcnt (no VGPR bounce).
static __device__ inline void async_copy_b128(const bf16* gsrc, bf16* lds_dst) {
  uint32_t lds_off = (uint32_t)(uintptr_t)lds_dst;  // addr[31:0] = LDS offset
  asm volatile("global_load_async_to_lds_b128 %0, %1, off"
               :: "v"(lds_off), "v"(gsrc) : "memory");
}
static __device__ inline void async_wait0() {
  asm volatile("s_wait_asynccnt 0x0" ::: "memory");
}

// ---------------------------------------------------------------------------
// fp32 -> bf16 conversion (hidden states)
// ---------------------------------------------------------------------------
__global__ void cvt_bf16_kernel(const float* __restrict__ in,
                                bf16* __restrict__ out, int n) {
  int i = blockIdx.x * 256 + threadIdx.x;
  if (i < n) out[i] = (bf16)in[i];
}

// ---------------------------------------------------------------------------
// Weight transpose+convert: Wt[n][k] (bf16) = W[k][n] (f32). 32x32 LDS tiles.
// ---------------------------------------------------------------------------
__global__ __launch_bounds__(256) void wtrans_kernel(const float* __restrict__ W,
                                                     bf16* __restrict__ Wt) {
  __shared__ float tile[32][33];
  const int k0 = blockIdx.x * 32;
  const int n0 = blockIdx.y * 32;
  const int tx = threadIdx.x & 31;
  const int ty0 = threadIdx.x >> 5;
#pragma unroll
  for (int i = 0; i < 4; ++i) {
    int ty = ty0 + i * 8;
    tile[ty][tx] = W[(size_t)(k0 + ty) * CH + n0 + tx];
  }
  __syncthreads();
#pragma unroll
  for (int i = 0; i < 4; ++i) {
    int ty = ty0 + i * 8;
    Wt[(size_t)(n0 + ty) * CH + k0 + tx] = (bf16)tile[tx][ty];
  }
}

// ---------------------------------------------------------------------------
// stage helpers (async b128 copies, entire workgroup participates)
// ---------------------------------------------------------------------------
static __device__ inline void stage_w_tile(const bf16* __restrict__ WtT,
                                           bf16* dst, int n0, int K, int k0,
                                           int tid) {
#pragma unroll
  for (int r = 0; r < 2; ++r) {
    int c    = r * 256 + tid;   // 512 chunks: 128 rows x 4 x 16B
    int n    = c >> 2;
    int koff = (c & 3) << 3;
    async_copy_b128(WtT + (size_t)(n0 + n) * K + k0 + koff,
                    dst + (size_t)n * 32 + koff);
  }
}

static __device__ inline void stage_v_tile(const bf16* __restrict__ vhead,
                                           bf16* dst, int j0, int tid) {
#pragma unroll
  for (int r = 0; r < 3; ++r) {
    int g = r * 256 + tid;      // 640 chunks: 160 rows x 4 x 16B
    if (g < 640) {
      int d    = g >> 2;
      int koff = (g & 3) << 3;
      async_copy_b128(vhead + (size_t)d * SEQ + j0 + koff,
                      dst + (size_t)d * 32 + koff);
    }
  }
}

// ---------------------------------------------------------------------------
// GEMM: out[M,N] = A[M,K](bf16) @ WtT[N,K](bf16, pre-transposed) (+bias)
// 256 threads = 8 waves; WG tile 64x128; wave tile 16x64.
// Double-buffered LDS W tiles staged with async b128 while WMMAs run.
// mode: 0 = bf16 row-major out, 1 = bf16 scatter into V^T[b][h][d][s],
//       2 = f32 out + bias
// ---------------------------------------------------------------------------
__global__ __launch_bounds__(256) void gemm_kernel(
    const bf16* __restrict__ A, const bf16* __restrict__ WtT,
    bf16* __restrict__ outb, float* __restrict__ outf,
    const float* __restrict__ bias, int mode, int M, int N, int K) {
  __shared__ __align__(16) bf16 Wt[2][128 * 32];  // [buf][n][k], 16 KB
  const int tid  = threadIdx.x;
  const int wid  = tid >> 5;
  const int lane = tid & 31;
  const int m0    = blockIdx.x * 64 + (wid & 3) * 16;
  const int n0    = blockIdx.y * 128;
  const int nside = (wid >> 2) * 64;

  v8f acc[4];
#pragma unroll
  for (int t = 0; t < 4; ++t) acc[t] = v8f_zero();

  const int nk = K / 32;
  stage_w_tile(WtT, &Wt[0][0], n0, K, 0, tid);  // prologue prefetch

  for (int ki = 0; ki < nk; ++ki) {
    async_wait0();     // my chunks of tile ki landed
    __syncthreads();   // everyone's chunks landed
    if (ki + 1 < nk)   // prefetch next tile while computing this one
      stage_w_tile(WtT, &Wt[(ki + 1) & 1][0], n0, K, (ki + 1) * 32, tid);

    const bf16* cur = &Wt[ki & 1][0];
    bf16x16 af = load_afrag(A + (size_t)m0 * K + (size_t)ki * 32, K);
    bf16x16 bfrag[4];
#pragma unroll
    for (int t = 0; t < 4; ++t)
      bfrag[t] = load_bfrag(cur + (size_t)(nside + t * 16) * 32, 32);
#pragma unroll
    for (int t = 0; t < 4; ++t) acc[t] = wmma_bf16(af, bfrag[t], acc[t]);

    __syncthreads();   // all reads of this buffer done before restage (ki+2)
  }

  const int ncol = lane & 15;
  const int half = lane >> 4;
#pragma unroll
  for (int t = 0; t < 4; ++t) {
    int n = n0 + nside + t * 16 + ncol;
#pragma unroll
    for (int r = 0; r < 8; ++r) {
      int m = m0 + r + 8 * half;
      if (mode == 0) {
        outb[(size_t)m * N + n] = (bf16)acc[t][r];
      } else if (mode == 1) {  // scatter into V^T[b][h][d][s]
        int h = n / HDIM, d = n % HDIM;
        int bb = m >> 12, s = m & (SEQ - 1);
        outb[(((size_t)bb * HEADS + h) * HDIM + d) * SEQ + s] = (bf16)acc[t][r];
      } else {
        outf[(size_t)m * N + n] = acc[t][r] + bias[n];
      }
    }
  }
}

// ---------------------------------------------------------------------------
// Flash attention: 256 threads = 8 waves; wave owns 16 query rows (128/WG).
// Double-buffered V^T tiles staged via async b128 (shared by 8 waves);
// K-fragments straight from global (+prefetch of next tile); online softmax
// via wave32 shuffles; P tile D->A layout bounce through per-wave LDS.
// ---------------------------------------------------------------------------
__global__ __launch_bounds__(256) void attn_kernel(
    const bf16* __restrict__ Q, const bf16* __restrict__ Km,
    const bf16* __restrict__ Vt_g, bf16* __restrict__ Ao) {
  __shared__ __align__(16) bf16 Vt[2][HDIM * 32];  // [buf][d][k], 20 KB
  __shared__ __align__(16) bf16 Pt[8][16 * 32];    // per-wave [m][k], 8 KB
  const int tid  = threadIdx.x;
  const int wid  = tid >> 5;
  const int lane = tid & 31;
  const int half = lane >> 4;
  const int b  = blockIdx.z;
  const int h  = blockIdx.y;
  const int q0 = blockIdx.x * 128 + wid * 16;
  const size_t headoff = (size_t)h * HDIM;
  const bf16* vhead = Vt_g + ((size_t)b * HEADS + h) * HDIM * SEQ;  // [d][s]

  const bf16* qbase = Q + ((size_t)b * SEQ + q0) * CH + headoff;
  bf16x16 qf[5];
#pragma unroll
  for (int c = 0; c < 5; ++c) qf[c] = load_afrag(qbase + c * 32, CH);

  float mrow[8], lrow[8];
#pragma unroll
  for (int r = 0; r < 8; ++r) { mrow[r] = -3.0e38f; lrow[r] = 0.0f; }
  v8f acc[10];
#pragma unroll
  for (int d = 0; d < 10; ++d) acc[d] = v8f_zero();

  const float scale = 0.07905694150420949f;  // 1/sqrt(160)

  stage_v_tile(vhead, &Vt[0][0], 0, tid);  // prologue prefetch

  for (int j0 = 0; j0 < SEQ; j0 += 32) {
    async_wait0();
    __syncthreads();
    if (j0 + 32 < SEQ)
      stage_v_tile(vhead, &Vt[((j0 >> 5) + 1) & 1][0], j0 + 32, tid);

    // scores S = Q @ K^T for two 16-key tiles
    const bf16* kbase = Km + ((size_t)b * SEQ + j0) * CH + headoff;
    if (j0 + 32 < SEQ)  // prefetch next K tile rows (global_prefetch_b8)
      __builtin_prefetch(kbase + (size_t)(32 + lane) * CH, 0, 3);
    v8f s[2];
#pragma unroll
    for (int t = 0; t < 2; ++t) {
      v8f sc = v8f_zero();
#pragma unroll
      for (int c = 0; c < 5; ++c) {
        bf16x16 kf = load_bfrag(kbase + (size_t)(t * 16) * CH + c * 32, CH);
        sc = wmma_bf16(qf[c], kf, sc);
      }
      s[t] = sc * scale;
    }

    // online softmax: reductions within the 16-lane group holding one row
    float alpha[8];
#pragma unroll
    for (int r = 0; r < 8; ++r) {
      float v = fmaxf(s[0][r], s[1][r]);
      v = fmaxf(v, __shfl_xor(v, 1, 32));
      v = fmaxf(v, __shfl_xor(v, 2, 32));
      v = fmaxf(v, __shfl_xor(v, 4, 32));
      v = fmaxf(v, __shfl_xor(v, 8, 32));
      float newm = fmaxf(mrow[r], v);
      float p0 = __expf(s[0][r] - newm);
      float p1 = __expf(s[1][r] - newm);
      s[0][r] = p0; s[1][r] = p1;
      float sum = p0 + p1;
      sum += __shfl_xor(sum, 1, 32);
      sum += __shfl_xor(sum, 2, 32);
      sum += __shfl_xor(sum, 4, 32);
      sum += __shfl_xor(sum, 8, 32);
      alpha[r] = __expf(mrow[r] - newm);
      lrow[r]  = lrow[r] * alpha[r] + sum;
      mrow[r]  = newm;
    }
#pragma unroll
    for (int d = 0; d < 10; ++d)
#pragma unroll
      for (int r = 0; r < 8; ++r) acc[d][r] *= alpha[r];

    // D-layout probs -> A-layout bf16 via per-wave LDS bounce
    bf16* pb = &Pt[wid][0];
#pragma unroll
    for (int t = 0; t < 2; ++t)
#pragma unroll
      for (int r = 0; r < 8; ++r)
        pb[(size_t)(r + 8 * half) * 32 + t * 16 + (lane & 15)] = (bf16)s[t][r];
    asm volatile("s_wait_dscnt 0x0" ::: "memory");  // same-wave LDS RAW fence

    bf16x16 pf = load_afrag(pb, 32);
    const bf16* vcur = &Vt[(j0 >> 5) & 1][0];
    bf16x16 vf[10];
#pragma unroll
    for (int d = 0; d < 10; ++d)
      vf[d] = load_bfrag(vcur + (size_t)(d * 16) * 32, 32);
#pragma unroll
    for (int d = 0; d < 10; ++d) acc[d] = wmma_bf16(pf, vf[d], acc[d]);

    __syncthreads();  // all reads of this V buffer done before restage
  }

  float inv[8];
#pragma unroll
  for (int r = 0; r < 8; ++r) inv[r] = 1.0f / lrow[r];
  const int ncol = lane & 15;
#pragma unroll
  for (int d = 0; d < 10; ++d)
#pragma unroll
    for (int r = 0; r < 8; ++r) {
      size_t row = (size_t)b * SEQ + q0 + r + 8 * half;
      Ao[row * CH + headoff + d * 16 + ncol] = (bf16)(acc[d][r] * inv[r]);
    }
}

// ---------------------------------------------------------------------------
// launch
// ---------------------------------------------------------------------------
extern "C" void kernel_launch(void* const* d_in, const int* in_sizes, int n_in,
                              void* d_out, int out_size, void* d_ws, size_t ws_size,
                              hipStream_t stream) {
  const float* hs = (const float*)d_in[0];
  const float* wq = (const float*)d_in[1];
  const float* wk = (const float*)d_in[2];
  const float* wv = (const float*)d_in[3];
  const float* wo = (const float*)d_in[4];
  const float* bo = (const float*)d_in[5];

  const size_t nTok   = (size_t)BATCH * SEQ;  // 8192
  const size_t nElem  = nTok * CH;            // 10,485,760
  const size_t bytesB = nElem * sizeof(bf16); // 20.97 MB
  const size_t wBytes = (size_t)CH * CH * sizeof(bf16);  // 3.28 MB

  char* ws = (char*)d_ws;
  bf16* Xb   = (bf16*)(ws + 0 * bytesB);
  bf16* Qb   = (bf16*)(ws + 1 * bytesB);
  bf16* Kb   = (bf16*)(ws + 2 * bytesB);
  bf16* VtG  = (bf16*)(ws + 3 * bytesB);  // [b][h][d][s]
  bf16* Aob  = (bf16*)(ws + 4 * bytesB);
  char* wsw  = ws + 5 * bytesB;
  bf16* WqT  = (bf16*)(wsw + 0 * wBytes);
  bf16* WkT  = (bf16*)(wsw + 1 * wBytes);
  bf16* WvT  = (bf16*)(wsw + 2 * wBytes);
  bf16* WoT  = (bf16*)(wsw + 3 * wBytes);

  dim3 tgrid(CH / 32, CH / 32);  // 40x40
  wtrans_kernel<<<tgrid, 256, 0, stream>>>(wq, WqT);
  wtrans_kernel<<<tgrid, 256, 0, stream>>>(wk, WkT);
  wtrans_kernel<<<tgrid, 256, 0, stream>>>(wv, WvT);
  wtrans_kernel<<<tgrid, 256, 0, stream>>>(wo, WoT);

  cvt_bf16_kernel<<<dim3((unsigned)((nElem + 255) / 256)), 256, 0, stream>>>(
      hs, Xb, (int)nElem);

  dim3 ggrid(128, 10);  // 8192/64 x 1280/128
  gemm_kernel<<<ggrid, 256, 0, stream>>>(Xb, WqT, Qb, nullptr, nullptr, 0,
                                         (int)nTok, CH, CH);
  gemm_kernel<<<ggrid, 256, 0, stream>>>(Xb, WkT, Kb, nullptr, nullptr, 0,
                                         (int)nTok, CH, CH);
  gemm_kernel<<<ggrid, 256, 0, stream>>>(Xb, WvT, VtG, nullptr, nullptr, 1,
                                         (int)nTok, CH, CH);

  attn_kernel<<<dim3(SEQ / 128, HEADS, BATCH), 256, 0, stream>>>(Qb, Kb, VtG, Aob);

  gemm_kernel<<<ggrid, 256, 0, stream>>>(Aob, WoT, nullptr, (float*)d_out, bo, 2,
                                         (int)nTok, CH, CH);
}
